// CorrelationTorch_919123001766
// MI455X (gfx1250) — matrix operand order repaired and checked by reading, not compiled, not run on hardware
//
#include <hip/hip_runtime.h>
#include <hip/hip_bf16.h>

#define BATCH    4
#define CH       256
#define HH       96
#define WW       192
#define OD       9      // 2*max_disp+1
#define PADD     4
#define TW       64     // x1 pixels per workgroup
#define KC       64     // channel chunk staged in LDS
#define STRD     72     // LDS row stride in f16 (16B-aligned rows, good bank spread)
#define X2C      80     // staged x2 columns (w0-4 .. w0+75)
#define NTHREADS 256

typedef __attribute__((ext_vector_type(16))) _Float16 v16h;
typedef __attribute__((ext_vector_type(8)))  float    v8f;
typedef __attribute__((ext_vector_type(4)))  unsigned int v4u;

union Frag {
  v16h h;
  v4u  q[2];
};

union Pack4 {
  _Float16 h[4];
  unsigned long long u64;
};

__device__ __forceinline__ void load_bfrag(Frag& f, const _Float16* __restrict__ sx2,
                                           int i, int cc, int k0, int nhalf) {
  const _Float16* base = &sx2[(i * X2C + cc) * STRD + k0 + 16 * nhalf];
  f.q[0] = *(const v4u*)(base);
  f.q[1] = *(const v4u*)(base + 8);
}

extern "C" __global__ __launch_bounds__(NTHREADS)
__attribute__((amdgpu_waves_per_eu(4)))   // LDS caps us at 4 waves/SIMD; free the VGPR budget
void corr_wmma_kernel(const float* __restrict__ x1,
                      const float* __restrict__ x2,
                      float* __restrict__ out)
{
  extern __shared__ _Float16 smem[];
  _Float16* sx1 = smem;                 // [TW][STRD]
  _Float16* sx2 = smem + TW * STRD;     // [OD][X2C][STRD]

  const int tid  = threadIdx.x;
  const int lane = tid & 31;            // wave32
  const int wave = tid >> 5;
  const int mt   = wave & 3;            // x1 16-pixel sub-tile
  const int nt   = wave >> 2;           // x2 16-col N-tile (0 or 1)

  int bx = blockIdx.x;
  const int wt = bx % (WW / TW);  bx /= (WW / TW);
  const int h  = bx % HH;
  const int b  = bx / HH;
  const int w0 = wt * TW;

  const size_t HW = (size_t)HH * WW;

  v8f acc[OD] = {};

  const int nhalf = lane >> 4;
  const int mlo   = lane & 15;
  const int cc    = 16 * (mt + nt) + mlo;   // B-fragment column index in staged x2

  for (int kc = 0; kc < CH; kc += KC) {
    __syncthreads();

    // ---- stage x1 chunk: global [c][w] (f32) -> LDS [w][c] (f16)
    {
      const float* g = x1 + ((size_t)(b * CH + kc) * HH + h) * WW + w0;
      for (int idx = tid; idx < TW * (KC / 4); idx += NTHREADS) {
        const int w = idx % TW;        // consecutive lanes -> consecutive w (coalesced)
        const int c = (idx / TW) * 4;
        Pack4 p;
        #pragma unroll
        for (int q = 0; q < 4; ++q)
          p.h[q] = (_Float16)g[(size_t)(c + q) * HW + w];
        *(unsigned long long*)(&sx1[w * STRD + c]) = p.u64;
      }
    }

    // ---- stage x2 chunk: 9 rows x 80 cols, zero-padded at borders
    {
      for (int idx = tid; idx < OD * X2C * (KC / 4); idx += NTHREADS) {
        const int c2 = idx % X2C;      // consecutive lanes -> consecutive w2
        const int t  = idx / X2C;
        const int c  = (t % (KC / 4)) * 4;
        const int r  = t / (KC / 4);
        const int h2 = h + r - PADD;
        const int w2 = w0 - PADD + c2;
        Pack4 p;
        if ((unsigned)h2 < (unsigned)HH && (unsigned)w2 < (unsigned)WW) {
          const float* g = x2 + ((size_t)(b * CH + kc + c) * HH + h2) * WW + w2;
          #pragma unroll
          for (int q = 0; q < 4; ++q)
            p.h[q] = (_Float16)g[(size_t)q * HW];
        } else {
          #pragma unroll
          for (int q = 0; q < 4; ++q) p.h[q] = (_Float16)0.0f;
        }
        *(unsigned long long*)(&sx2[(r * X2C + c2) * STRD + c]) = p.u64;
      }
    }
    __syncthreads();

    // ---- WMMA: 2 k-steps of 32 channels; preload A + all 9 B fragments,
    //      then drain through the matrix pipe (loads stay in flight).
    #pragma unroll
    for (int ks = 0; ks < KC / 32; ++ks) {
      const int k0 = ks * 32;

      // A fragment: x1 pixels 16*mt..16*mt+15, contiguous k order (matches B, not the ISA table):
      // lane<16: K 0..15; lane>=16: K 16..31 (contiguous halves)
      Frag a;
      {
        const _Float16* base = &sx1[(16 * mt + mlo) * STRD + k0 + 16 * nhalf];
        a.q[0] = *(const v4u*)(base);        // K = 16h .. 16h+7
        a.q[1] = *(const v4u*)(base + 8);    // K = 16h+8 .. 16h+15
      }

      // B fragments: lane<16: K 0..15 ; lane>=16: K 16..31 (contiguous halves)
      Frag bb[OD];
      #pragma unroll
      for (int i = 0; i < OD; ++i)
        load_bfrag(bb[i], sx2, i, cc, k0, nhalf);

      #pragma unroll
      for (int i = 0; i < OD; ++i)
        acc[i] = __builtin_amdgcn_wmma_f32_16x16x32_f16(
            false, a.h, false, bb[i].h, (short)0, acc[i], false, false);
    }
  }

  // ---- scatter banded results: D[m][n] -> out[b][i*9+j][h][w], j = 16*nt + n - m
  const float scale = 1.0f / (float)CH;
  #pragma unroll
  for (int i = 0; i < OD; ++i) {
    #pragma unroll
    for (int v = 0; v < 8; ++v) {
      const int m = v + 8 * nhalf;         // D-layout: VGPR v, lane-half -> M
      const int j = 16 * nt + mlo - m;     // mlo == N
      if (0 <= j && j <= 8) {
        const int w = w0 + 16 * mt + m;
        const size_t o = ((size_t)(b * (OD * OD) + i * OD + j) * HH + h) * WW + w;
        out[o] = acc[i][v] * scale;
      }
    }
  }
}

extern "C" void kernel_launch(void* const* d_in, const int* in_sizes, int n_in,
                              void* d_out, int out_size, void* d_ws, size_t ws_size,
                              hipStream_t stream) {
  (void)in_sizes; (void)n_in; (void)out_size; (void)d_ws; (void)ws_size;
  const float* x1 = (const float*)d_in[0];
  const float* x2 = (const float*)d_in[1];
  float* out = (float*)d_out;

  const dim3 grid(BATCH * HH * (WW / TW));   // 4*96*3 = 1152 workgroups
  const size_t shmem = (size_t)(TW * STRD + OD * X2C * STRD) * sizeof(_Float16); // 112,896 B
  corr_wmma_kernel<<<grid, NTHREADS, shmem, stream>>>(x1, x2, out);
}
